// MultiHeadAttention_5884105195732
// MI455X (gfx1250) — matrix operand order repaired and checked by reading, not compiled, not run on hardware
//
#include <hip/hip_runtime.h>
#include <stdint.h>

#define DEV __device__ __forceinline__

typedef __attribute__((ext_vector_type(16))) _Float16 v16h;
typedef __attribute__((ext_vector_type(8)))  _Float16 v8h;
typedef __attribute__((ext_vector_type(8)))  float    v8f;

constexpr int Bc = 2, Sc = 2048, Dc = 1024, Hc = 16, Dh = 64;
constexpr float NEGF = -3.4028234663852886e38f;   // finfo(float32).min

// ---------------------------------------------------------------------------
// WMMA helper: D = A(16x32 f16) x B(32x16 f16) + C(16x16 f32)
DEV v8f wmma32(v16h a, v16h b, v8f c) {
  return __builtin_amdgcn_wmma_f32_16x16x32_f16(
      /*neg_a=*/false, a, /*neg_b=*/false, b,
      /*c_mod=*/(short)0, c, /*reuse_a=*/false, /*reuse_b=*/false);
}

// Async DMA: 16 bytes global -> LDS, tracked by ASYNCcnt (CDNA5).
// LDS operand is the low 32 bits of the generic pointer (LDS aperture rule).
DEV void async_copy_b128(uint32_t lds_byte_addr, const void* src) {
  asm volatile("global_load_async_to_lds_b128 %0, %1, off"
               :: "v"(lds_byte_addr), "v"(src) : "memory");
}
DEV void wait_async0() {
  asm volatile("s_wait_asynccnt 0x0" ::: "memory");
}

// A-fragment (16x32, f16) from row-major storage with leading dim `ld`.
// Lane r=L&15, hi=L>>4; halves 0..7 -> K=hi*8+h, halves 8..15 -> K=16+hi*8+h.
DEV v16h load_a_frag(const _Float16* p, int ld) {
  const int lane = threadIdx.x & 31;
  const int r = lane & 15, hi = lane >> 4;
  const _Float16* row = p + (size_t)r * ld + hi * 8;
  v16h a;
#pragma unroll
  for (int h = 0; h < 8; ++h) a[h] = row[h];
#pragma unroll
  for (int h = 0; h < 8; ++h) a[8 + h] = row[16 + h];
  return a;
}

// B-fragment (32x16, f16) where B[k][n] = W[n][k], W row-major, ld = stride.
// Lane: n=L&15; halves 0..7 -> K=hi*8+h, halves 8..15 -> K=16+hi*8+h.
DEV v16h load_bT_frag(const _Float16* p, int ld) {
  const int lane = threadIdx.x & 31;
  const int n = lane & 15, hi = lane >> 4;
  const _Float16* row = p + (size_t)n * ld + hi * 8;
  v16h b;
#pragma unroll
  for (int h = 0; h < 8; ++h) b[h] = row[h];
#pragma unroll
  for (int h = 0; h < 8; ++h) b[8 + h] = row[16 + h];
  return b;
}

// ---------------------------------------------------------------------------
// Y[M,N] = ((A[M,K] @ W[N,K]^T) + bias[N]) * oscale
// Block: 256 threads (8 waves). Tile: 256(M) x 64(N); wave owns 32x64
// (2 A-frags, 8 WMMAs per 32-wide K step).
template <typename AT, typename OT>
__global__ __launch_bounds__(256) void gemm_xwT_bias(
    const AT* __restrict__ A, const float* __restrict__ W,
    const float* __restrict__ bias, OT* __restrict__ Y,
    int M, int N, int K, float oscale) {
  __shared__ _Float16 As[256 * 32];
  __shared__ _Float16 Ws[64 * 32];
  const int tid  = threadIdx.x;
  const int wave = tid >> 5;
  const int lane = tid & 31;
  const int n0 = blockIdx.x * 64;
  const int m0 = blockIdx.y * 256;

  v8f acc[2][4] = {};

  for (int k0 = 0; k0 < K; k0 += 32) {
#pragma unroll
    for (int i = tid; i < 256 * 32; i += 256) {
      int r = i >> 5, c = i & 31;
      As[i] = (_Float16)A[(size_t)(m0 + r) * K + k0 + c];
    }
#pragma unroll
    for (int i = tid; i < 64 * 32; i += 256) {
      int r = i >> 5, c = i & 31;
      Ws[i] = (_Float16)W[(size_t)(n0 + r) * K + k0 + c];
    }
    __syncthreads();

    v16h a0 = load_a_frag(As + (wave * 32) * 32, 32);
    v16h a1 = load_a_frag(As + (wave * 32 + 16) * 32, 32);
#pragma unroll
    for (int nt = 0; nt < 4; ++nt) {
      v16h bF = load_bT_frag(Ws + nt * 16 * 32, 32);
      acc[0][nt] = wmma32(a0, bF, acc[0][nt]);
      acc[1][nt] = wmma32(a1, bF, acc[1][nt]);
    }
    __syncthreads();
  }

  const int hi8  = (lane >> 4) << 3;
  const int col0 = n0 + (lane & 15);
#pragma unroll
  for (int mh = 0; mh < 2; ++mh) {
    const int row0 = m0 + wave * 32 + mh * 16 + hi8;
#pragma unroll
    for (int nt = 0; nt < 4; ++nt) {
      const int col = col0 + nt * 16;
      const float bv = bias[col];
#pragma unroll
      for (int v = 0; v < 8; ++v) {
        Y[(size_t)(row0 + v) * N + col] = (OT)((acc[mh][nt][v] + bv) * oscale);
      }
    }
  }
}

// ---------------------------------------------------------------------------
// Flash attention over one (b, h). Block: 256 thr (8 waves) = 128 q rows.
// Each wave: 16 q rows, streams 32 keys/iteration.
// S^T = K_tile @ Q^T so the two 16x16 f32 accumulators map exactly onto the
// A-fragment of P (16x32) -- no cross-lane shuffle; softmax stats per-lane.
// Q was pre-scaled by 1/sqrt(Dh) in its projection epilogue.
// K tiles: double-buffered global_load_async_to_lds_b128 (ASYNCcnt) prefetch.
// V tiles: prefetched to VGPRs, stored transposed into LDS.
__global__ __launch_bounds__(256) void flash_attn(
    const _Float16* __restrict__ Qp, const _Float16* __restrict__ Kp,
    const _Float16* __restrict__ Vp, const int* __restrict__ mask,
    _Float16* __restrict__ Cc) {
  __shared__ _Float16 Ks[2][32 * 64];  // [key][d] row-major
  __shared__ _Float16 Vt[2][64 * 32];  // [d][key] transposed
  const int tid  = threadIdx.x;
  const int wave = tid >> 5;
  const int lane = tid & 31;
  const int b = blockIdx.y / Hc;
  const int h = blockIdx.y % Hc;
  const int q0  = blockIdx.x * 128 + wave * 16;
  const int nq  = lane & 15;            // q column in S^T layout
  const int hi8 = (lane >> 4) << 3;

  const _Float16* Qbase = Qp + ((size_t)b * Sc + q0) * Dc + h * Dh;
  const _Float16* Kbase = Kp + ((size_t)b * Sc) * Dc + h * Dh;
  const _Float16* Vbase = Vp + ((size_t)b * Sc) * Dc + h * Dh;
  const int* mrow = mask + (size_t)(q0 + nq) * Sc;

  // Staging role of this thread: one 16B segment of one key row per tile.
  const int srow = tid >> 3;  // 0..31  (key row in tile)
  const int sseg = tid & 7;   // 0..7   (16B segment within 128B row)
  const _Float16* kSrc = Kbase + (size_t)srow * Dc + sseg * 8;
  const _Float16* vSrc = Vbase + (size_t)srow * Dc + sseg * 8;

  // Q as B-operand of S^T: B[d][q] = Q[q][d]; held in registers for all kt.
  const v16h bQ0 = load_bT_frag(Qbase, Dc);
  const v16h bQ1 = load_bT_frag(Qbase + 32, Dc);

  v8f acc[4] = {};
  float mrun = NEGF;
  float lrun = 0.f;

  // Prologue: prefetch tile 0 into buffer 0.
  async_copy_b128((uint32_t)(uintptr_t)&Ks[0][srow * 64 + sseg * 8], kSrc);
  v8h vv = *(const v8h*)vSrc;

  for (int kt = 0; kt < Sc; kt += 32) {
    const int buf = (kt >> 5) & 1;

    wait_async0();                       // this wave's K DMA has landed
#pragma unroll
    for (int j = 0; j < 8; ++j)          // transpose-store V tile
      Vt[buf][(sseg * 8 + j) * 32 + srow] = vv[j];
    __syncthreads();                     // all waves' K + V visible

    if (kt + 32 < Sc) {                  // prefetch next tile into other buf
      const int nb = buf ^ 1;
      async_copy_b128((uint32_t)(uintptr_t)&Ks[nb][srow * 64 + sseg * 8],
                      kSrc + (size_t)(kt + 32) * Dc);
      vv = *(const v8h*)(vSrc + (size_t)(kt + 32) * Dc);
    }

    const _Float16* K0 = Ks[buf];
    v16h aK00 = load_a_frag(K0,                64);  // keys +0..15,  d 0..31
    v16h aK01 = load_a_frag(K0 + 32,           64);  // keys +0..15,  d 32..63
    v16h aK10 = load_a_frag(K0 + 16 * 64,      64);  // keys +16..31, d 0..31
    v16h aK11 = load_a_frag(K0 + 16 * 64 + 32, 64);

    v8f z = {};
    v8f st0 = wmma32(aK00, bQ0, z);
    st0     = wmma32(aK01, bQ1, st0);
    v8f st1 = wmma32(aK10, bQ0, z);
    st1     = wmma32(aK11, bQ1, st1);

    // S^T element v: score(q = q0+nq, key = kt + t*16 + hi8 + v)
    // (Q pre-scaled, so WMMA output is already the scaled score.)
    float s0[8], s1[8];
#pragma unroll
    for (int v = 0; v < 8; ++v) {
      s0[v] = mrow[kt + hi8 + v]      ? st0[v] : NEGF;
      s1[v] = mrow[kt + 16 + hi8 + v] ? st1[v] : NEGF;
    }

    // Online softmax: per-lane over 16 values + one shfl_xor(16) combine.
    float pm = NEGF;
#pragma unroll
    for (int v = 0; v < 8; ++v) pm = fmaxf(pm, fmaxf(s0[v], s1[v]));
    pm = fmaxf(pm, __shfl_xor(pm, 16, 32));
    const float mnew = fmaxf(mrun, pm);
    const float corr = __expf(mrun - mnew);

    float rs = 0.f;
    v16h aP;  // A-fragment of P(16q x 32k): halves 0..7 <- st0, 8..15 <- st1
#pragma unroll
    for (int v = 0; v < 8; ++v) {
      float p0 = __expf(s0[v] - mnew);
      float p1 = __expf(s1[v] - mnew);
      rs += p0 + p1;
      aP[v]     = (_Float16)p0;
      aP[8 + v] = (_Float16)p1;
    }
    rs += __shfl_xor(rs, 16, 32);
    lrun = lrun * corr + rs;
    mrun = mnew;

    // Rescale accumulators; out row q_local = hi8+v, corr for q' is in lane q'.
#pragma unroll
    for (int v = 0; v < 8; ++v) {
      float cv = __shfl(corr, hi8 + v, 32);
#pragma unroll
      for (int dt = 0; dt < 4; ++dt) acc[dt][v] *= cv;
    }

    // out += P @ V : B[k][d] = V[kt+k][d] = Vt[d][k]
#pragma unroll
    for (int dt = 0; dt < 4; ++dt) {
      v16h bV = load_bT_frag(Vt[buf] + dt * 16 * 32, 32);
      acc[dt] = wmma32(aP, bV, acc[dt]);
    }
    __syncthreads();                     // done reading this buffer
  }

  const float linv = 1.0f / lrun;
#pragma unroll
  for (int v = 0; v < 8; ++v) {
    const float sv = __shfl(linv, hi8 + v, 32);
    const int rowg = q0 + hi8 + v;
    _Float16* orow = Cc + ((size_t)b * Sc + rowg) * Dc + h * Dh;
#pragma unroll
    for (int dt = 0; dt < 4; ++dt) {
      orow[dt * 16 + (lane & 15)] = (_Float16)(acc[dt][v] * sv);
    }
  }
}

// ---------------------------------------------------------------------------
extern "C" void kernel_launch(void* const* d_in, const int* in_sizes, int n_in,
                              void* d_out, int out_size, void* d_ws, size_t ws_size,
                              hipStream_t stream) {
  (void)in_sizes; (void)n_in; (void)out_size; (void)ws_size;
  const float* q    = (const float*)d_in[0];
  const float* k    = (const float*)d_in[1];
  const float* v    = (const float*)d_in[2];
  const int*   mask = (const int*)d_in[3];
  const float* Wq   = (const float*)d_in[4];
  const float* bq   = (const float*)d_in[5];
  const float* Wk   = (const float*)d_in[6];
  const float* bk   = (const float*)d_in[7];
  const float* Wv   = (const float*)d_in[8];
  const float* bv   = (const float*)d_in[9];
  const float* Wo   = (const float*)d_in[10];
  const float* bo   = (const float*)d_in[11];
  float* out = (float*)d_out;

  const int M = Bc * Sc;          // 4096
  const int N = Dc, K = Dc;       // 1024
  const size_t NE = (size_t)M * Dc;
  const float qscale = 0.125f;    // 1/sqrt(Dh), folded into Q projection

  _Float16* qp = (_Float16*)d_ws;  // f16 activations: 4 x 8MB, L2-resident
  _Float16* kp = qp + NE;
  _Float16* vp = kp + NE;
  _Float16* cc = vp + NE;

  dim3 blk(256);
  dim3 gg(N / 64, M / 256);
  gemm_xwT_bias<float, _Float16><<<gg, blk, 0, stream>>>(q, Wq, bq, qp, M, N, K, qscale);
  gemm_xwT_bias<float, _Float16><<<gg, blk, 0, stream>>>(k, Wk, bk, kp, M, N, K, 1.0f);
  gemm_xwT_bias<float, _Float16><<<gg, blk, 0, stream>>>(v, Wv, bv, vp, M, N, K, 1.0f);

  dim3 ga(Sc / 128, Bc * Hc);
  flash_attn<<<ga, blk, 0, stream>>>(qp, kp, vp, mask, cc);

  gemm_xwT_bias<_Float16, float><<<gg, blk, 0, stream>>>(cc, Wo, bo, out, M, N, K, 1.0f);
}
